// Transformer_70497593197177
// MI455X (gfx1250) — matrix operand order repaired and checked by reading, not compiled, not run on hardware
//
#include <hip/hip_runtime.h>
#include <math.h>

// ---------------------------------------------------------------------------
// CDNA5 (gfx1250) transformer forward. wave32, WMMA f32_16x16x32_f16.
// ---------------------------------------------------------------------------

typedef __attribute__((ext_vector_type(16))) _Float16 v16h;
typedef __attribute__((ext_vector_type(8)))  float    v8f;

union FragU { v16h v; unsigned u[8]; };

__device__ __forceinline__ v8f wmma_f16(v16h a, v16h b, v8f c) {
  return __builtin_amdgcn_wmma_f32_16x16x32_f16(
      false, a, false, b, (short)0, c, false, false);
}

// Pack float4 -> 4 f16 and store as one 8-byte LDS store (dst must be 8B aligned).
__device__ __forceinline__ void store4h(_Float16* dst, float4 v) {
  union { _Float16 h[4]; uint2 u; } p;
  p.h[0] = (_Float16)v.x; p.h[1] = (_Float16)v.y;
  p.h[2] = (_Float16)v.z; p.h[3] = (_Float16)v.w;
  *(uint2*)dst = p.u;
}

// A fragment (16x32 f16) from LDS tile stored row-major [m][k], strideH halfs (even).
// ISA 7.12.2 layout: lanes 0-15 -> M, VGPR v<4: K=2v,2v+1 (+8 for lanes 16-31),
// VGPR v>=4: K=16+2(v-4)(+8) ; K pairs contiguous -> dword loads.
__device__ __forceinline__ v16h load_frag_a(const _Float16* tile, int strideH,
                                            int mBase, int kOfs) {
  const int lane = threadIdx.x & 31;
  const int hl = lane >> 4;
  const unsigned* row = (const unsigned*)(tile + (size_t)(mBase + (lane & 15)) * strideH);
  FragU f;
#pragma unroll
  for (int vg = 0; vg < 8; ++vg) {
    int kb = (vg < 4) ? (2 * vg + 8 * hl) : (16 + 2 * (vg - 4) + 8 * hl);
    f.u[vg] = row[(kOfs + kb) >> 1];
  }
  return f.v;
}

// B fragment (32x16 f16) from LDS tile stored transposed [n][k], strideH halfs (even).
// Layout: lane -> N, same K order as A: VGPR v<4: K=2v,2v+1 (+8 for lanes 16-31), v>=4: K=16+2(v-4)(+8).
__device__ __forceinline__ v16h load_frag_b(const _Float16* tile, int strideH,
                                            int nBase, int kOfs) {
  const int lane = threadIdx.x & 31;
  const int hl = lane >> 4;
  const unsigned* row = (const unsigned*)(tile + (size_t)(nBase + (lane & 15)) * strideH);
  FragU f;
#pragma unroll
  for (int vg = 0; vg < 8; ++vg) {
    int k = kOfs + ((vg < 4) ? (2 * vg + 8 * hl) : (16 + 2 * (vg - 4) + 8 * hl));
    f.u[vg] = row[k >> 1];
  }
  return f.v;
}

// ---------------------------------------------------------------------------
// Generic GEMM: C = act(A[MxK] @ B[KxN] + bias), fp32 in/out, f16 WMMA inside.
// Requirements (guaranteed by caller): M % 128 == 0, K % 32 == 0. N arbitrary.
// Block tile 128x128, 256 threads (8 waves as 4m x 2n), wave tile 32x64.
// Software-pipelined global->reg->LDS staging. act: 0 none, 1 exact GELU.
// ---------------------------------------------------------------------------
#define AST 40
#define BST 40

__global__ __launch_bounds__(256) void gemm_kernel(
    const float* __restrict__ A, const float* __restrict__ B,
    const float* __restrict__ bias, float* __restrict__ C,
    int M, int N, int K, int act) {
  __shared__ __align__(16) _Float16 As[128 * AST];
  __shared__ __align__(16) _Float16 Bs[128 * BST];
  const int tid = threadIdx.x;
  const int lane = tid & 31, wid = tid >> 5;
  const int hl = lane >> 4, ln15 = lane & 15;
  const int wm = wid >> 1, wn = wid & 1;
  const int m0 = blockIdx.x * 128, n0 = blockIdx.y * 128;

  v8f c[2][4];
#pragma unroll
  for (int i = 0; i < 2; ++i)
#pragma unroll
    for (int j = 0; j < 4; ++j)
#pragma unroll
      for (int v = 0; v < 8; ++v) c[i][j][v] = 0.0f;

  float4 aR[4];   // A tile 128x32 = 1024 float4 / 256 thr
  float  bR[16];  // B tile 32x128 = 4096 elems / 256 thr (row-contig reads)

  auto loadA = [&](int k0) {
#pragma unroll
    for (int i = 0; i < 4; ++i) {
      int idx4 = tid + 256 * i;
      int r = idx4 >> 3, cc = (idx4 & 7) * 4;
      aR[i] = *(const float4*)&A[(size_t)(m0 + r) * K + k0 + cc];
    }
  };
  auto loadB = [&](int k0) {
#pragma unroll
    for (int i = 0; i < 16; ++i) {
      int idx = tid + 256 * i;
      int n = idx & 127, k = idx >> 7;
      int gn = n0 + n;
      int gnc = gn < N ? gn : N - 1;           // clamp: unconditional load
      float v = B[(size_t)(k0 + k) * N + gnc];
      bR[i] = gn < N ? v : 0.0f;
    }
  };

  loadA(0);
  loadB(0);
  for (int k0 = 0; k0 < K; k0 += 32) {
    __syncthreads();
#pragma unroll
    for (int i = 0; i < 4; ++i) {
      int idx4 = tid + 256 * i;
      int r = idx4 >> 3, cc = (idx4 & 7) * 4;
      store4h(&As[r * AST + cc], aR[i]);
    }
#pragma unroll
    for (int i = 0; i < 16; ++i) {
      int idx = tid + 256 * i;
      int n = idx & 127, k = idx >> 7;
      Bs[n * BST + k] = (_Float16)bR[i];
    }
    __syncthreads();
    if (k0 + 32 < K) {  // prefetch next tile; overlaps with WMMAs below
      loadA(k0 + 32);
      loadB(k0 + 32);
    }
    v16h a0 = load_frag_a(As, AST, wm * 32, 0);
    v16h a1 = load_frag_a(As, AST, wm * 32 + 16, 0);
#pragma unroll
    for (int j = 0; j < 4; ++j) {
      v16h bf = load_frag_b(Bs, BST, wn * 64 + j * 16, 0);
      c[0][j] = wmma_f16(a0, bf, c[0][j]);
      c[1][j] = wmma_f16(a1, bf, c[1][j]);
    }
  }

#pragma unroll
  for (int i = 0; i < 2; ++i)
#pragma unroll
    for (int j = 0; j < 4; ++j) {
      int nn = n0 + wn * 64 + j * 16 + ln15;
      if (nn < N) {
#pragma unroll
        for (int v = 0; v < 8; ++v) {
          int mm = m0 + wm * 32 + i * 16 + v + 8 * hl;  // M%128==0 -> in range
          float val = c[i][j][v];
          if (bias) val += bias[nn];
          if (act == 1) val = 0.5f * val * (1.0f + erff(val * 0.70710678118654752f));
          C[(size_t)mm * N + nn] = val;
        }
      }
    }
}

// ---------------------------------------------------------------------------
// Flash attention: qkv [4096, 3072] fp32 -> out [4096, 1024] fp32 (b,n,h*d).
// grid (16 row-blocks, 64 b*h), 128 threads (4 waves, 16 query rows each).
// K/V tile staging pipelined across j-steps.
// ---------------------------------------------------------------------------
#define FST 72
#define VST 40

__global__ __launch_bounds__(128) void flash_kernel(
    const float* __restrict__ qkv, float* __restrict__ out) {
  __shared__ __align__(16) _Float16 Qt[64 * FST];
  __shared__ __align__(16) _Float16 Kt[32 * FST];
  __shared__ __align__(16) _Float16 Vt[64 * VST];
  __shared__ __align__(16) _Float16 Ps[4 * 16 * 32];

  const int tid = threadIdx.x, lane = tid & 31, wid = tid >> 5;
  const int hl = lane >> 4, ln15 = lane & 15;
  const int ib = blockIdx.x;
  const int bh = blockIdx.y;
  const int b = bh >> 4, h = bh & 15;
  const size_t rowbase = (size_t)b * 1024;
  const int i0 = ib * 64;
  const float SCALE = 0.125f;  // 64^-0.5

#pragma unroll
  for (int i = 0; i < 8; ++i) {  // Q tile 64x64 = 1024 float4
    int idx4 = tid + 128 * i;
    int r = idx4 >> 4, cc = (idx4 & 15) * 4;
    float4 qv = *(const float4*)&qkv[(rowbase + i0 + r) * 3072 + h * 64 + cc];
    store4h(&Qt[r * FST + cc], qv);
  }
  __syncthreads();
  v16h aq0 = load_frag_a(Qt, FST, wid * 16, 0);
  v16h aq1 = load_frag_a(Qt, FST, wid * 16, 32);

  float mrun[8], lrun[8];
  v8f o[4];
#pragma unroll
  for (int v = 0; v < 8; ++v) { mrun[v] = -1e30f; lrun[v] = 0.0f; }
#pragma unroll
  for (int f = 0; f < 4; ++f)
#pragma unroll
    for (int v = 0; v < 8; ++v) o[f][v] = 0.0f;

  _Float16* Pw = Ps + wid * 512;

  float4 Kr[4], Vr[4];  // K/V tiles 32x64 = 512 float4 each / 128 thr
  auto loadKV = [&](int j0) {
#pragma unroll
    for (int i = 0; i < 4; ++i) {
      int idx4 = tid + 128 * i;
      int r = idx4 >> 4, cc = (idx4 & 15) * 4;
      const float* base = &qkv[(rowbase + j0 + r) * 3072 + h * 64 + cc];
      Kr[i] = *(const float4*)(base + 1024);
      Vr[i] = *(const float4*)(base + 2048);
    }
  };

  loadKV(0);
  for (int j0 = 0; j0 < 1024; j0 += 32) {
    __syncthreads();
#pragma unroll
    for (int i = 0; i < 4; ++i) {
      int idx4 = tid + 128 * i;
      int r = idx4 >> 4, cc = (idx4 & 15) * 4;
      store4h(&Kt[r * FST + cc], Kr[i]);           // K: [j'][d]  (= B [n][k])
      const float* vf = &Vr[i].x;
#pragma unroll
      for (int q = 0; q < 4; ++q)                   // V transposed -> [d][j']
        Vt[(cc + q) * VST + r] = (_Float16)vf[q];
    }
    __syncthreads();
    if (j0 + 32 < 1024) loadKV(j0 + 32);  // prefetch next tile

    v8f s0, s1;
#pragma unroll
    for (int v = 0; v < 8; ++v) { s0[v] = 0.0f; s1[v] = 0.0f; }
    {
      v16h bk = load_frag_b(Kt, FST, 0, 0);
      s0 = wmma_f16(aq0, bk, s0);
      bk = load_frag_b(Kt, FST, 0, 32);
      s0 = wmma_f16(aq1, bk, s0);
      bk = load_frag_b(Kt, FST, 16, 0);
      s1 = wmma_f16(aq0, bk, s1);
      bk = load_frag_b(Kt, FST, 16, 32);
      s1 = wmma_f16(aq1, bk, s1);
    }

#pragma unroll
    for (int v = 0; v < 8; ++v) {
      float a0 = s0[v] * SCALE, a1 = s1[v] * SCALE;
      float mx = fmaxf(a0, a1);
      for (int off = 1; off < 16; off <<= 1)
        mx = fmaxf(mx, __shfl_xor(mx, off, 32));
      float mn = fmaxf(mrun[v], mx);
      float corr = __expf(mrun[v] - mn);
      float p0 = __expf(a0 - mn), p1 = __expf(a1 - mn);
      float ps = p0 + p1;
      for (int off = 1; off < 16; off <<= 1) ps += __shfl_xor(ps, off, 32);
      mrun[v] = mn;
      lrun[v] = lrun[v] * corr + ps;
#pragma unroll
      for (int f = 0; f < 4; ++f) o[f][v] *= corr;
      Pw[(v + 8 * hl) * 32 + ln15] = (_Float16)p0;
      Pw[(v + 8 * hl) * 32 + 16 + ln15] = (_Float16)p1;
    }

    v16h ap = load_frag_a(Pw, 32, 0, 0);
#pragma unroll
    for (int nt = 0; nt < 4; ++nt) {
      v16h bv = load_frag_b(Vt, VST, nt * 16, 0);
      o[nt] = wmma_f16(ap, bv, o[nt]);
    }
  }

#pragma unroll
  for (int f = 0; f < 4; ++f)
#pragma unroll
    for (int v = 0; v < 8; ++v) {
      int r = i0 + wid * 16 + v + 8 * hl;
      int col = h * 64 + f * 16 + ln15;
      out[(rowbase + r) * 1024 + col] = o[f][v] / lrun[v];
    }
}

// ---------------------------------------------------------------------------
// Last-layer dots: attn[bh][i][j] = SCALE * Q[i,:] . K[j,:]  (NT stores)
// grid (16 i-tiles, 16 j-tiles, 64 bh), 128 threads.
// ---------------------------------------------------------------------------
__global__ __launch_bounds__(128) void qk_kernel(
    const float* __restrict__ qkv, float* __restrict__ attn) {
  __shared__ __align__(16) _Float16 Qt[64 * FST];
  __shared__ __align__(16) _Float16 Kt[64 * FST];
  const int tid = threadIdx.x, lane = tid & 31, wid = tid >> 5;
  const int hl = lane >> 4, ln15 = lane & 15;
  const int it = blockIdx.x, jt = blockIdx.y, bh = blockIdx.z;
  const int b = bh >> 4, h = bh & 15;
  const size_t rowbase = (size_t)b * 1024;
  const int i0 = it * 64, j0 = jt * 64;

#pragma unroll
  for (int i = 0; i < 8; ++i) {
    int idx4 = tid + 128 * i;
    int r = idx4 >> 4, cc = (idx4 & 15) * 4;
    float4 qv = *(const float4*)&qkv[(rowbase + i0 + r) * 3072 + h * 64 + cc];
    float4 kv = *(const float4*)&qkv[(rowbase + j0 + r) * 3072 + 1024 + h * 64 + cc];
    store4h(&Qt[r * FST + cc], qv);
    store4h(&Kt[r * FST + cc], kv);
  }
  __syncthreads();

  v16h aq0 = load_frag_a(Qt, FST, wid * 16, 0);
  v16h aq1 = load_frag_a(Qt, FST, wid * 16, 32);
  float* dst = attn + (size_t)bh * 1024 * 1024;

#pragma unroll
  for (int nt = 0; nt < 4; ++nt) {
    v8f c;
#pragma unroll
    for (int v = 0; v < 8; ++v) c[v] = 0.0f;
    c = wmma_f16(aq0, load_frag_b(Kt, FST, nt * 16, 0), c);
    c = wmma_f16(aq1, load_frag_b(Kt, FST, nt * 16, 32), c);
#pragma unroll
    for (int v = 0; v < 8; ++v) {
      int r = i0 + wid * 16 + v + 8 * hl;
      int cc = j0 + nt * 16 + ln15;
      __builtin_nontemporal_store(c[v] * 0.125f, &dst[(size_t)r * 1024 + cc]);
    }
  }
}

// ---------------------------------------------------------------------------
// Row softmax over 1024 elements, in place, single NT read + NT write.
// grid = 65536 rows.
// ---------------------------------------------------------------------------
__global__ __launch_bounds__(256) void softmax_kernel(float* __restrict__ p) {
  __shared__ float red[256];
  float* r = p + (size_t)blockIdx.x * 1024;
  const int tid = threadIdx.x;
  float vals[4];
  float mx = -1e30f;
#pragma unroll
  for (int j = 0; j < 4; ++j) {
    vals[j] = __builtin_nontemporal_load(&r[tid + 256 * j]);
    mx = fmaxf(mx, vals[j]);
  }
  red[tid] = mx;
  __syncthreads();
  for (int s = 128; s > 0; s >>= 1) {
    if (tid < s) red[tid] = fmaxf(red[tid], red[tid + s]);
    __syncthreads();
  }
  mx = red[0];
  __syncthreads();
  float sum = 0.0f;
#pragma unroll
  for (int j = 0; j < 4; ++j) {
    vals[j] = __expf(vals[j] - mx);
    sum += vals[j];
  }
  red[tid] = sum;
  __syncthreads();
  for (int s = 128; s > 0; s >>= 1) {
    if (tid < s) red[tid] += red[tid + s];
    __syncthreads();
  }
  float inv = 1.0f / red[0];
#pragma unroll
  for (int j = 0; j < 4; ++j)
    __builtin_nontemporal_store(vals[j] * inv, &r[tid + 256 * j]);
}

// ---------------------------------------------------------------------------
// LayerNorm (+ optional residual): out = LN(a)*g + b (+ resid). grid = rows.
// ---------------------------------------------------------------------------
__global__ __launch_bounds__(256) void ln_kernel(
    const float* __restrict__ a, const float* __restrict__ g,
    const float* __restrict__ beta, const float* __restrict__ resid,
    float* __restrict__ out, int d) {
  __shared__ float red[256];
  const int tid = threadIdx.x;
  const size_t base = (size_t)blockIdx.x * d;
  float s = 0.0f, s2 = 0.0f;
  for (int i = tid; i < d; i += 256) {
    float v = a[base + i];
    s += v;
    s2 += v * v;
  }
  red[tid] = s;
  __syncthreads();
  for (int t = 128; t > 0; t >>= 1) {
    if (tid < t) red[tid] += red[tid + t];
    __syncthreads();
  }
  float mean = red[0] / (float)d;
  __syncthreads();
  red[tid] = s2;
  __syncthreads();
  for (int t = 128; t > 0; t >>= 1) {
    if (tid < t) red[tid] += red[tid + t];
    __syncthreads();
  }
  float var = red[0] / (float)d - mean * mean;
  float rs = rsqrtf(var + 1e-5f);
  for (int i = tid; i < d; i += 256) {
    float v = (a[base + i] - mean) * rs * g[i] + beta[i];
    if (resid) v += resid[base + i];
    out[base + i] = v;
  }
}

// ---------------------------------------------------------------------------
// Host orchestration
// ---------------------------------------------------------------------------
extern "C" void kernel_launch(void* const* d_in, const int* in_sizes, int n_in,
                              void* d_out, int out_size, void* d_ws, size_t ws_size,
                              hipStream_t stream) {
  (void)in_sizes; (void)n_in; (void)out_size; (void)ws_size;
  const int D[4] = {1024, 512, 256, 128};
  const int M = 4096;

  int idx = 0;
  const float* X0 = (const float*)d_in[idx++];
  const float *Wqkv[4], *Wo[4], *bo[4], *lg1[4], *lb1[4], *lg2[4], *lb2[4];
  const float *W1[4], *fb1[4], *W2[4], *fb2[4];
  for (int l = 0; l < 4; ++l) {
    Wqkv[l] = (const float*)d_in[idx++];
    Wo[l]   = (const float*)d_in[idx++];
    bo[l]   = (const float*)d_in[idx++];
    lg1[l]  = (const float*)d_in[idx++];
    lb1[l]  = (const float*)d_in[idx++];
    if (l < 3) {
      lg2[l] = (const float*)d_in[idx++];
      lb2[l] = (const float*)d_in[idx++];
    } else {
      lg2[l] = nullptr;
      lb2[l] = nullptr;
    }
    W1[l]  = (const float*)d_in[idx++];
    fb1[l] = (const float*)d_in[idx++];
    W2[l]  = (const float*)d_in[idx++];
    fb2[l] = (const float*)d_in[idx++];
  }

  char* w = (char*)d_ws;
  float* B0 = (float*)(w);                                  // x       16 MB
  float* Bq = (float*)(w + (size_t)16 * 1024 * 1024);       // qkv/t   48 MB
  float* Bt = (float*)(w + (size_t)64 * 1024 * 1024);       // ao/h    16 MB
  float* Ba = (float*)(w + (size_t)80 * 1024 * 1024);       // a       16 MB
  float* out = (float*)d_out;
  float* attn = out + 12288;

  const float* xsrc = X0;
  for (int l = 0; l < 4; ++l) {
    int d = D[l], hid = d * 3 / 4, dn = (l < 3) ? d / 2 : 3;

    // 1. qkv = x @ Wqkv                       [M, 3072]
    gemm_kernel<<<dim3(M / 128, 3072 / 128), 256, 0, stream>>>(
        xsrc, Wqkv[l], nullptr, Bq, M, 3072, d, 0);

    // 2. flash attention -> concat-head output [M, 1024]
    flash_kernel<<<dim3(16, 64), 128, 0, stream>>>(Bq, Bt);

    // Last layer: materialize attn probs into d_out.
    if (l == 3) {
      qk_kernel<<<dim3(16, 16, 64), 128, 0, stream>>>(Bq, attn);
      softmax_kernel<<<dim3(65536), 256, 0, stream>>>(attn);
    }

    // 3. a = ao @ Wo + bo                     [M, d]
    gemm_kernel<<<dim3(M / 128, (d + 127) / 128), 256, 0, stream>>>(
        Bt, Wo[l], bo[l], Ba, M, d, 1024, 0);

    // 4. x = LN(a)*g1 + b1 + x               [M, d] -> B0
    ln_kernel<<<4096, 256, 0, stream>>>(Ba, lg1[l], lb1[l], xsrc, B0, d);

    // 5. h = LN(x) (layers 0-2) else x
    const float* hsrc = B0;
    if (l < 3) {
      ln_kernel<<<4096, 256, 0, stream>>>(B0, lg2[l], lb2[l], nullptr, Bt, d);
      hsrc = Bt;
    }

    // 6. t = gelu(h @ W1 + b1)               [M, hid] -> Bq (qkv consumed)
    gemm_kernel<<<dim3(M / 128, (hid + 127) / 128), 256, 0, stream>>>(
        hsrc, W1[l], fb1[l], Bq, M, hid, d, 1);

    // 7. x = t @ W2 + b2                     [M, dn]
    float* xdst = (l == 3) ? out : B0;
    gemm_kernel<<<dim3(M / 128, (dn + 127) / 128), 256, 0, stream>>>(
        Bq, W2[l], fb2[l], xdst, M, dn, hid, 0);

    xsrc = B0;
  }
}
